// FlashLinks_79267916415444
// MI455X (gfx1250) — matrix operand order repaired and checked by reading, not compiled, not run on hardware
//
#include <hip/hip_runtime.h>
#include <hip/hip_bf16.h>
#include <stdint.h>

#define B_ 8
#define S_ 1024
#define H_ 512
#define D_ 64
#define NH_ 2
#define EPS_ 1e-6f

typedef __attribute__((ext_vector_type(16))) __bf16 v16bf;
typedef __attribute__((ext_vector_type(8)))  float  v8f;
typedef __attribute__((ext_vector_type(4)))  unsigned int v4u;

union Frag32B { v16bf v; v4u q[2]; };

__device__ __forceinline__ unsigned short f2bf(float f){
  unsigned int u = __float_as_uint(f);
  u += 0x7FFFu + ((u >> 16) & 1u);
  return (unsigned short)(u >> 16);
}
__device__ __forceinline__ float bf2f(unsigned short u){
  return __uint_as_float(((unsigned int)u) << 16);
}
__device__ __forceinline__ v8f zero8(){
  v8f z = {0.f,0.f,0.f,0.f,0.f,0.f,0.f,0.f};
  return z;
}

// A-fragment (16x32 bf16, MxK): lane<16 holds row M=lane, K = k0..k0+7 & k0+16..k0+23;
// lane>=16 holds row M=lane-16, K = k0+8..k0+15 & k0+24..k0+31 (ISA 7.12.2).
__device__ __forceinline__ v16bf load_a_frag(const unsigned short* __restrict__ P,
                                             int ld, int m0, int k0, int lane){
  Frag32B f;
  const unsigned short* p = P + (size_t)(m0 + (lane & 15)) * ld + (k0 + ((lane >> 4) << 3));
  f.q[0] = *(const v4u*)p;
  f.q[1] = *(const v4u*)(p + 16);
  return f.v;
}
// B-fragment (32x16 bf16, KxN) from a "QT" buffer where row n == column n of B:
// lane<16 holds col N=lane, K = k0..k0+7 & k0+16..k0+23; lane>=16 holds K = k0+8..k0+15 & k0+24..k0+31.
__device__ __forceinline__ v16bf load_bt_frag(const unsigned short* __restrict__ QT,
                                              int ld, int n0, int k0, int lane){
  Frag32B f;
  const unsigned short* p = QT + (size_t)(n0 + (lane & 15)) * ld + (k0 + ((lane >> 4) << 3));
  f.q[0] = *(const v4u*)p;
  f.q[1] = *(const v4u*)(p + 16);
  return f.v;
}
__device__ __forceinline__ v8f wmma_bf16(v16bf a, v16bf b, v8f c){
  return __builtin_amdgcn_wmma_f32_16x16x32_bf16(false, a, false, b, (short)0, c, false, false);
}

// ---------------- converts ----------------
__global__ void k_f32_to_bf16(const float* __restrict__ in, unsigned short* __restrict__ out, int n){
  int i = blockIdx.x * blockDim.x + threadIdx.x;
  if (i < n) out[i] = f2bf(in[i]);
}

// ---------------- projection GEMM: Out[M,64] = Xbf[M,512] @ Wbf[64,512]^T ----------------
__global__ void k_proj_gemm(const unsigned short* __restrict__ Xbf,
                            const unsigned short* __restrict__ Wbf,
                            float* __restrict__ Out){
  int lane = threadIdx.x;
  int m0 = blockIdx.x * 16;
  v8f acc[4] = {zero8(), zero8(), zero8(), zero8()};
  for (int k0 = 0; k0 < H_; k0 += 32){
    v16bf a = load_a_frag(Xbf, H_, m0, k0, lane);
    #pragma unroll
    for (int j = 0; j < 4; ++j){
      v16bf b = load_bt_frag(Wbf, H_, j * 16, k0, lane);
      acc[j] = wmma_bf16(a, b, acc[j]);
    }
  }
  int rl_hi = (lane >> 4) << 3;
  int cl = lane & 15;
  #pragma unroll
  for (int r = 0; r < 8; ++r){
    int row = m0 + r + rl_hi;
    #pragma unroll
    for (int j = 0; j < 4; ++j)
      Out[(size_t)row * D_ + j * 16 + cl] = acc[j][r];
  }
}

// ---------------- RoPE + bf16 pack ----------------
__global__ void k_rope(const float* __restrict__ Qf, const float* __restrict__ Kf,
                       unsigned short* __restrict__ Qbf, unsigned short* __restrict__ Kbf){
  int i = blockIdx.x * blockDim.x + threadIdx.x;
  if (i >= B_ * S_ * D_) return;
  int d = i & 63;
  int s = (i >> 6) & (S_ - 1);
  float inv = __powf(10000.f, -(float)(d & 31) * (2.f / 64.f));
  float ang = (float)s * inv;
  float c = __cosf(ang), sn = __sinf(ang);
  float qr = (d < 32) ? -Qf[i + 32] : Qf[i - 32];
  float kr = (d < 32) ? -Kf[i + 32] : Kf[i - 32];
  Qbf[i] = f2bf(Qf[i] * c + qr * sn);
  Kbf[i] = f2bf(Kf[i] * c + kr * sn);
}

// ---------------- scores (WMMA) + bias + temperature + softmax -> A, A^T ----------------
__global__ void k_attn(const unsigned short* __restrict__ Qbf, const unsigned short* __restrict__ Kbf,
                       const float* __restrict__ bias_table, const int* __restrict__ rel_idx,
                       const float* __restrict__ temperature,
                       unsigned short* __restrict__ Abf, unsigned short* __restrict__ AbfT){
  __shared__ float smem[16 * S_];     // 64 KB
  int lane = threadIdx.x & 31;
  int wave = threadIdx.x >> 5;        // 8 waves
  int m0 = blockIdx.x * 16;
  int b  = blockIdx.y;
  const unsigned short* Qb = Qbf + (size_t)b * S_ * D_;
  const unsigned short* Kb = Kbf + (size_t)b * S_ * D_;
  float tcl = fmaxf(temperature[0], 0.1f);
  float inv_st = 1.0f / (8.0f * tcl);           // 1/(sqrt(64)*temp)
  float inv_t  = 1.0f / tcl;
  v16bf a0 = load_a_frag(Qb, D_, m0, 0, lane);
  v16bf a1 = load_a_frag(Qb, D_, m0, 32, lane);
  int rl_hi = (lane >> 4) << 3;
  int cl = lane & 15;
  for (int nt = wave; nt < S_ / 16; nt += 8){
    int n0 = nt * 16;
    v16bf b0 = load_bt_frag(Kb, D_, n0, 0, lane);
    v16bf b1 = load_bt_frag(Kb, D_, n0, 32, lane);
    v8f acc = zero8();
    acc = wmma_bf16(a0, b0, acc);
    acc = wmma_bf16(a1, b1, acc);
    #pragma unroll
    for (int r = 0; r < 8; ++r){
      int rl = r + rl_hi;
      int sg = m0 + rl;
      int tg = n0 + cl;
      float pb = bias_table[rel_idx[(size_t)sg * S_ + tg]];
      smem[rl * S_ + tg] = acc[r] * inv_st + pb * inv_t;
    }
  }
  __syncthreads();
  size_t base = (size_t)b * S_ * S_;
  for (int rr = 0; rr < 2; ++rr){
    int rl = wave * 2 + rr;
    float* row = smem + rl * S_;
    float m = -3.0e38f;
    for (int t = lane; t < S_; t += 32) m = fmaxf(m, row[t]);
    #pragma unroll
    for (int off = 16; off > 0; off >>= 1) m = fmaxf(m, __shfl_xor(m, off, 32));
    float sum = 0.f;
    for (int t = lane; t < S_; t += 32){
      float e = __expf(row[t] - m);
      row[t] = e;
      sum += e;
    }
    #pragma unroll
    for (int off = 16; off > 0; off >>= 1) sum += __shfl_xor(sum, off, 32);
    float invs = 1.0f / sum;
    int sg = m0 + rl;
    for (int t = lane; t < S_; t += 32){
      unsigned short u = f2bf(row[t] * invs);
      Abf [base + (size_t)sg * S_ + t] = u;
      AbfT[base + (size_t)t * S_ + sg] = u;
    }
  }
}

// ---------------- generic SxS bf16 WMMA GEMM: C = P @ Q (QT holds Q's columns as rows) ----
// 64x32 output tile per wave: 4 A-frags + 2 B-frags feed 8 WMMAs per k-step.
__global__ void k_gemm_ss(const unsigned short* __restrict__ P, const unsigned short* __restrict__ QT,
                          unsigned short* __restrict__ C){
  int lane = threadIdx.x;
  int m0 = blockIdx.x * 64;
  int n0 = blockIdx.y * 32;
  size_t base = (size_t)blockIdx.z * S_ * S_;
  const unsigned short* Pb = P + base;
  const unsigned short* Qb = QT + base;
  v8f acc[4][2];
  #pragma unroll
  for (int i = 0; i < 4; ++i)
    #pragma unroll
    for (int j = 0; j < 2; ++j) acc[i][j] = zero8();
  for (int k0 = 0; k0 < S_; k0 += 32){
    v16bf fb0 = load_bt_frag(Qb, S_, n0, k0, lane);
    v16bf fb1 = load_bt_frag(Qb, S_, n0 + 16, k0, lane);
    #pragma unroll
    for (int i = 0; i < 4; ++i){
      v16bf fa = load_a_frag(Pb, S_, m0 + i * 16, k0, lane);
      acc[i][0] = wmma_bf16(fa, fb0, acc[i][0]);
      acc[i][1] = wmma_bf16(fa, fb1, acc[i][1]);
    }
  }
  int rl_hi = (lane >> 4) << 3;
  int cl = lane & 15;
  unsigned short* Cb = C + base;
  #pragma unroll
  for (int i = 0; i < 4; ++i){
    #pragma unroll
    for (int r = 0; r < 8; ++r){
      int row = m0 + i * 16 + r + rl_hi;
      Cb[(size_t)row * S_ + n0 + cl]      = f2bf(acc[i][0][r]);
      Cb[(size_t)row * S_ + n0 + 16 + cl] = f2bf(acc[i][1][r]);
    }
  }
}

// ---------------- row sums of a [B*S, S] bf16 matrix ----------------
__global__ void k_rowsum(const unsigned short* __restrict__ C, float* __restrict__ rs){
  __shared__ float red[8];
  int tid = threadIdx.x;
  const unsigned short* row = C + (size_t)blockIdx.x * S_;
  float s = 0.f;
  for (int t = tid; t < S_; t += 256) s += bf2f(row[t]);
  #pragma unroll
  for (int off = 16; off > 0; off >>= 1) s += __shfl_xor(s, off, 32);
  if ((tid & 31) == 0) red[tid >> 5] = s;
  __syncthreads();
  if (tid == 0){
    float tot = 0.f;
    for (int w = 0; w < 8; ++w) tot += red[w];
    rs[blockIdx.x] = tot;
  }
}

// ---------------- fused prob-log + sigmoid + row sum ----------------
__device__ __forceinline__ float plog(float p){
  p = fminf(fmaxf(p, EPS_), 1.f - EPS_);
  return __logf(p) - __logf(1.f - p + EPS_);
}

__global__ void k_fused(const unsigned short* __restrict__ Abf,
                        const unsigned short* __restrict__ AAt,  const float* __restrict__ rsAAt,
                        const unsigned short* __restrict__ AtA,  const float* __restrict__ rsAtA,
                        const unsigned short* __restrict__ A2,   const float* __restrict__ rsA2,
                        const unsigned short* __restrict__ AAt2, const float* __restrict__ rsAAt2,
                        const unsigned short* __restrict__ AtA2, const float* __restrict__ rsAtA2,
                        const float* __restrict__ fusion_w, const float* __restrict__ fusion_b,
                        unsigned short* __restrict__ fusedRaw, float* __restrict__ rsF){
  __shared__ float red[8];
  int g = blockIdx.x;        // row over B*S
  int hop = blockIdx.y;
  int tid = threadIdx.x;
  const unsigned short *P1, *P2, *P3;
  float i1, i2, i3;
  if (hop == 0){
    P1 = Abf;  i1 = 1.f / (1.f + EPS_);           // softmax rows sum to 1
    P2 = AAt;  i2 = 1.f / (rsAAt[g] + EPS_);
    P3 = AtA;  i3 = 1.f / (rsAtA[g] + EPS_);
  } else {
    P1 = A2;   i1 = 1.f / (rsA2[g] + EPS_);
    P2 = AAt2; i2 = 1.f / (rsAAt2[g] + EPS_);
    P3 = AtA2; i3 = 1.f / (rsAtA2[g] + EPS_);
  }
  float w0 = fusion_w[hop * 3 + 0], w1 = fusion_w[hop * 3 + 1], w2 = fusion_w[hop * 3 + 2];
  float fb = fusion_b[hop];
  size_t rbase = (size_t)g * S_;
  unsigned short* out = fusedRaw + (size_t)hop * B_ * S_ * S_ + rbase;
  float loc = 0.f;
  for (int t = tid; t < S_; t += 256){
    float l = fb + w0 * plog(bf2f(P1[rbase + t]) * i1)
                 + w1 * plog(bf2f(P2[rbase + t]) * i2)
                 + w2 * plog(bf2f(P3[rbase + t]) * i3);
    float f = 1.f / (1.f + __expf(-l));
    out[t] = f2bf(f);
    loc += f;
  }
  #pragma unroll
  for (int off = 16; off > 0; off >>= 1) loc += __shfl_xor(loc, off, 32);
  if ((tid & 31) == 0) red[tid >> 5] = loc;
  __syncthreads();
  if (tid == 0){
    float tot = 0.f;
    for (int w = 0; w < 8; ++w) tot += red[w];
    rsF[hop * (B_ * S_) + g] = tot;
  }
}

// ---------------- hop_v transposed: hopvT[hop][b][e][s] = sum_d V[b,s,d] * hop_W[hop,e,d] ----------------
__global__ void k_hopv(const float* __restrict__ Vf, const float* __restrict__ hop_W,
                       unsigned short* __restrict__ hopvT){
  int i = blockIdx.x * blockDim.x + threadIdx.x;
  if (i >= NH_ * B_ * D_ * S_) return;
  int s   = i & (S_ - 1);
  int e   = (i >> 10) & (D_ - 1);
  int b   = (i >> 16) & (B_ - 1);
  int hop = i >> 19;
  const float* v = Vf + ((size_t)b * S_ + s) * D_;
  const float* w = hop_W + ((size_t)hop * D_ + e) * D_;
  float acc = 0.f;
  for (int d = 0; d < D_; ++d) acc += v[d] * w[d];
  hopvT[i] = f2bf(acc);
}

// ---------------- gate softmax ----------------
__global__ void k_gate(const float* __restrict__ x, const float* __restrict__ gate_W,
                       const float* __restrict__ gate_b, float* __restrict__ gate){
  int i = blockIdx.x * blockDim.x + threadIdx.x;
  if (i >= B_ * S_) return;
  const float* xr = x + (size_t)i * H_;
  float l0 = gate_b[0], l1 = gate_b[1];
  for (int h = 0; h < H_; ++h){
    float xv = xr[h];
    l0 += xv * gate_W[h];
    l1 += xv * gate_W[H_ + h];
  }
  float m = fmaxf(l0, l1);
  float e0 = __expf(l0 - m), e1 = __expf(l1 - m);
  float inv = 1.f / (e0 + e1);
  gate[i * 2 + 0] = e0 * inv;
  gate[i * 2 + 1] = e1 * inv;
}

// ---------------- combined = sum_h gate_h/rowsum_h * (fusedRaw_h @ hop_v_h) ----------------
__global__ void k_combine(const unsigned short* __restrict__ fusedRaw,
                          const unsigned short* __restrict__ hopvT,
                          const float* __restrict__ rsF, const float* __restrict__ gate,
                          unsigned short* __restrict__ combined){
  int lane = threadIdx.x;
  int m0 = blockIdx.x * 16;      // global row over B*S
  int b = m0 / S_;
  int mloc = m0 - b * S_;
  const size_t BSS = (size_t)B_ * S_ * S_;
  v8f acc[2][4];
  #pragma unroll
  for (int h = 0; h < 2; ++h)
    #pragma unroll
    for (int j = 0; j < 4; ++j) acc[h][j] = zero8();
  for (int k0 = 0; k0 < S_; k0 += 32){
    #pragma unroll
    for (int h = 0; h < 2; ++h){
      v16bf a = load_a_frag(fusedRaw + (size_t)h * BSS + (size_t)b * S_ * S_, S_, mloc, k0, lane);
      #pragma unroll
      for (int j = 0; j < 4; ++j){
        v16bf bb = load_bt_frag(hopvT + ((size_t)h * B_ + b) * D_ * S_, S_, j * 16, k0, lane);
        acc[h][j] = wmma_bf16(a, bb, acc[h][j]);
      }
    }
  }
  int rl_hi = (lane >> 4) << 3;
  int cl = lane & 15;
  #pragma unroll
  for (int r = 0; r < 8; ++r){
    int row = m0 + r + rl_hi;
    float g0 = gate[row * 2 + 0] / (rsF[row] + EPS_);
    float g1 = gate[row * 2 + 1] / (rsF[B_ * S_ + row] + EPS_);
    #pragma unroll
    for (int j = 0; j < 4; ++j){
      float v = acc[0][j][r] * g0 + acc[1][j][r] * g1;
      combined[(size_t)row * D_ + j * 16 + cl] = f2bf(v);
    }
  }
}

// ---------------- out = combined @ out_W^T ----------------
__global__ void k_out(const unsigned short* __restrict__ combined,
                      const unsigned short* __restrict__ outWbf,
                      float* __restrict__ out){
  int lane = threadIdx.x;
  int m0 = blockIdx.x * 16;
  int n0 = blockIdx.y * 64;
  v8f acc[4] = {zero8(), zero8(), zero8(), zero8()};
  for (int k0 = 0; k0 < D_; k0 += 32){
    v16bf a = load_a_frag(combined, D_, m0, k0, lane);
    #pragma unroll
    for (int j = 0; j < 4; ++j){
      v16bf b = load_bt_frag(outWbf, D_, n0 + j * 16, k0, lane);
      acc[j] = wmma_bf16(a, b, acc[j]);
    }
  }
  int rl_hi = (lane >> 4) << 3;
  int cl = lane & 15;
  #pragma unroll
  for (int r = 0; r < 8; ++r){
    int row = m0 + r + rl_hi;
    #pragma unroll
    for (int j = 0; j < 4; ++j)
      out[(size_t)row * H_ + n0 + j * 16 + cl] = acc[j][r];
  }
}

extern "C" void kernel_launch(void* const* d_in, const int* in_sizes, int n_in,
                              void* d_out, int out_size, void* d_ws, size_t ws_size,
                              hipStream_t stream){
  (void)in_sizes; (void)n_in; (void)out_size; (void)ws_size;
  const float* x      = (const float*)d_in[0];
  const float* Wq     = (const float*)d_in[1];
  const float* Wk     = (const float*)d_in[2];
  const float* Wv     = (const float*)d_in[3];
  const float* biasT  = (const float*)d_in[4];
  const float* fw     = (const float*)d_in[5];
  const float* fb     = (const float*)d_in[6];
  const float* hopW   = (const float*)d_in[7];
  const float* gateW  = (const float*)d_in[8];
  const float* gateb  = (const float*)d_in[9];
  const float* outW   = (const float*)d_in[10];
  const float* temp   = (const float*)d_in[11];
  const int*   relidx = (const int*)d_in[12];
  float* out = (float*)d_out;

  char* ws = (char*)d_ws;
  size_t off = 0;
  auto alloc = [&](size_t bytes) -> char* {
    char* p = ws + off;
    off += (bytes + 255) & ~(size_t)255;
    return p;
  };
  const size_t BS  = (size_t)B_ * S_;
  const size_t BSS = BS * S_;

  unsigned short* xbf    = (unsigned short*)alloc(BS * H_ * 2);
  unsigned short* Wqbf   = (unsigned short*)alloc((size_t)D_ * H_ * 2);
  unsigned short* Wkbf   = (unsigned short*)alloc((size_t)D_ * H_ * 2);
  unsigned short* Wvbf   = (unsigned short*)alloc((size_t)D_ * H_ * 2);
  unsigned short* outWbf = (unsigned short*)alloc((size_t)H_ * D_ * 2);
  float* Qf = (float*)alloc(BS * D_ * 4);
  float* Kf = (float*)alloc(BS * D_ * 4);
  float* Vf = (float*)alloc(BS * D_ * 4);
  unsigned short* Qbf  = (unsigned short*)alloc(BS * D_ * 2);
  unsigned short* Kbf  = (unsigned short*)alloc(BS * D_ * 2);
  unsigned short* Abf  = (unsigned short*)alloc(BSS * 2);
  unsigned short* AbfT = (unsigned short*)alloc(BSS * 2);
  unsigned short* AAt  = (unsigned short*)alloc(BSS * 2);
  unsigned short* AtA  = (unsigned short*)alloc(BSS * 2);
  unsigned short* A2   = (unsigned short*)alloc(BSS * 2);
  unsigned short* AAt2 = (unsigned short*)alloc(BSS * 2);
  unsigned short* AtA2 = (unsigned short*)alloc(BSS * 2);
  float* rsAAt  = (float*)alloc(BS * 4);
  float* rsAtA  = (float*)alloc(BS * 4);
  float* rsA2   = (float*)alloc(BS * 4);
  float* rsAAt2 = (float*)alloc(BS * 4);
  float* rsAtA2 = (float*)alloc(BS * 4);
  unsigned short* fusedRaw = (unsigned short*)alloc((size_t)NH_ * BSS * 2);
  float* rsF = (float*)alloc((size_t)NH_ * BS * 4);
  unsigned short* hopvT = (unsigned short*)alloc((size_t)NH_ * B_ * D_ * S_ * 2);
  float* gateBuf = (float*)alloc(BS * 2 * 4);
  unsigned short* combined = (unsigned short*)alloc(BS * D_ * 2);

  int n;
  n = (int)(BS * H_); k_f32_to_bf16<<<(n + 255) / 256, 256, 0, stream>>>(x, xbf, n);
  n = D_ * H_;        k_f32_to_bf16<<<(n + 255) / 256, 256, 0, stream>>>(Wq, Wqbf, n);
  n = D_ * H_;        k_f32_to_bf16<<<(n + 255) / 256, 256, 0, stream>>>(Wk, Wkbf, n);
  n = D_ * H_;        k_f32_to_bf16<<<(n + 255) / 256, 256, 0, stream>>>(Wv, Wvbf, n);
  n = H_ * D_;        k_f32_to_bf16<<<(n + 255) / 256, 256, 0, stream>>>(outW, outWbf, n);

  k_proj_gemm<<<dim3((unsigned)(BS / 16)), 32, 0, stream>>>(xbf, Wqbf, Qf);
  k_proj_gemm<<<dim3((unsigned)(BS / 16)), 32, 0, stream>>>(xbf, Wkbf, Kf);
  k_proj_gemm<<<dim3((unsigned)(BS / 16)), 32, 0, stream>>>(xbf, Wvbf, Vf);

  n = (int)(BS * D_); k_rope<<<(n + 255) / 256, 256, 0, stream>>>(Qf, Kf, Qbf, Kbf);

  k_attn<<<dim3(S_ / 16, B_), 256, 0, stream>>>(Qbf, Kbf, biasT, relidx, temp, Abf, AbfT);

  dim3 gss(S_ / 64, S_ / 32, B_);
  k_gemm_ss<<<gss, 32, 0, stream>>>(Abf,  Abf,  AAt);   // A @ A^T (symmetric)
  k_gemm_ss<<<gss, 32, 0, stream>>>(AbfT, AbfT, AtA);   // A^T @ A (symmetric)
  k_gemm_ss<<<gss, 32, 0, stream>>>(Abf,  AbfT, A2);    // A @ A
  k_rowsum<<<(unsigned)BS, 256, 0, stream>>>(AAt, rsAAt);
  k_rowsum<<<(unsigned)BS, 256, 0, stream>>>(AtA, rsAtA);
  k_rowsum<<<(unsigned)BS, 256, 0, stream>>>(A2,  rsA2);
  k_gemm_ss<<<gss, 32, 0, stream>>>(AAt, AAt, AAt2);    // (AAt)^2, symmetric operand
  k_gemm_ss<<<gss, 32, 0, stream>>>(AtA, AtA, AtA2);    // (AtA)^2, symmetric operand
  k_rowsum<<<(unsigned)BS, 256, 0, stream>>>(AAt2, rsAAt2);
  k_rowsum<<<(unsigned)BS, 256, 0, stream>>>(AtA2, rsAtA2);

  k_fused<<<dim3((unsigned)BS, NH_), 256, 0, stream>>>(Abf, AAt, rsAAt, AtA, rsAtA,
                                                       A2, rsA2, AAt2, rsAAt2, AtA2, rsAtA2,
                                                       fw, fb, fusedRaw, rsF);

  n = NH_ * B_ * D_ * S_; k_hopv<<<(n + 255) / 256, 256, 0, stream>>>(Vf, hopW, hopvT);
  n = (int)BS;            k_gate<<<(n + 255) / 256, 256, 0, stream>>>(x, gateW, gateb, gateBuf);

  k_combine<<<dim3((unsigned)(BS / 16)), 32, 0, stream>>>(fusedRaw, hopvT, rsF, gateBuf, combined);
  k_out<<<dim3((unsigned)(BS / 16), H_ / 64), 32, 0, stream>>>(combined, outWbf, out);
}
